// GatedDeltaNet_16690242912586
// MI455X (gfx1250) — matrix operand order repaired and checked by reading, not compiled, not run on hardware
//
#include <hip/hip_runtime.h>
#include <hip/hip_bf16.h>
#include <math.h>

// ---------------------------------------------------------------------------
// Gated DeltaNet forward for MI455X (gfx1250, wave32, WMMA + async-LDS).
// B=4, L=1024, DH=2048, D=128, HQ=16, HK=HV=4, H=16, Kconv=4.
// ---------------------------------------------------------------------------

typedef __attribute__((ext_vector_type(16))) __bf16 bf16x16;
typedef __attribute__((ext_vector_type(8)))  float  f32x8;

#define B_    4
#define L_    1024
#define DH_   2048
#define D_    128
#define HQ_   16
#define HK_   4
#define HV_   4
#define H_    16
#define KC_   4
#define ML_   (B_ * L_)          // 4096 rows
#define HD_   (H_ * D_)          // 2048

__device__ __forceinline__ float sigmoidf_(float x) {
    return 1.0f / (1.0f + __expf(-x));
}
__device__ __forceinline__ float siluf_(float x) {
    return x * sigmoidf_(x);
}

// LDS byte offset of a __shared__ object: generic LDS pointers carry the
// wave-relative LDS offset in their low 32 bits (aperture mapping).
template <typename T>
__device__ __forceinline__ uint32_t lds_off_(T* p) {
    return (uint32_t)(uintptr_t)p;
}

// Async copy of 16 bytes global -> LDS (per lane), tracked by ASYNCcnt.
__device__ __forceinline__ void async_ld_b128_(uint32_t lds_addr, const void* g) {
    asm volatile("global_load_async_to_lds_b128 %0, %1, off"
                 :: "v"(lds_addr), "v"(g) : "memory");
}
__device__ __forceinline__ void wait_asynccnt0_() {
    asm volatile("s_wait_asynccnt 0x0" ::: "memory");
}

// ---------------------------------------------------------------------------
// Kernel 0: f32 -> bf16 conversion (n % 4 == 0)
// ---------------------------------------------------------------------------
__global__ __launch_bounds__(256)
void to_bf16_kernel(const float* __restrict__ src, __bf16* __restrict__ dst, int n) {
    const int i = (blockIdx.x * 256 + threadIdx.x) * 4;
    if (i + 3 < n) {
        #pragma unroll
        for (int j = 0; j < 4; ++j) dst[i + j] = (__bf16)src[i + j];
    }
}

// ---------------------------------------------------------------------------
// Kernel 1: GEMM with bf16 A, f32 W, f32 C.
//   C[M,N] = A[M,K] * W[K,N]   (row-major).  Requires M%128==0, N%64==0, K%32==0,
//   and K*2 bytes row stride of A divisible by 16 (true: K=2048).
// Block = 256 threads (8 waves).  Tile 128(M) x 64(N), K-step 32.
// A tile staged with GLOBAL_LOAD_ASYNC_TO_LDS_B128 (ASYNCcnt);
// B tile converted f32->bf16 and stored transposed so both fragment
// reads are contiguous 128-bit LDS loads.
// ---------------------------------------------------------------------------
__global__ __launch_bounds__(256)
void wmma_gemm_kernel(const __bf16* __restrict__ A, const float* __restrict__ W,
                      float* __restrict__ C, int M, int N, int Kdim) {
    __shared__ __bf16 As[128][32];   // [m][k]    8 KB  (async-filled)
    __shared__ __bf16 Bs[64][32];    // [n][k]    4 KB  (transposed)

    const int tid  = threadIdx.x;
    const int wid  = tid >> 5;
    const int lane = tid & 31;
    const int wm   = wid & 3;        // 0..3 -> M sub-tile of 32
    const int wn   = wid >> 2;       // 0..1 -> N sub-tile of 32
    const int hf   = lane >> 4;      // half-wave select
    const int l16  = lane & 15;

    const int m0 = blockIdx.y * 128;
    const int n0 = blockIdx.x * 64;

    // staging coordinates
    const int ar  = tid >> 1;              // A row 0..127
    const int ac  = (tid & 1) * 16;        // A col base (16 bf16 = 32 B)
    const int bk  = tid >> 3;              // B k-row 0..31
    const int bc  = (tid & 7) * 8;         // B col base (8 floats)

    const uint32_t asA0 = lds_off_(&As[ar][ac]);
    const uint32_t asA1 = lds_off_(&As[ar][ac + 8]);

    f32x8 acc[2][2] = {};

    for (int k0 = 0; k0 < Kdim; k0 += 32) {
        // ---- A tile: 2 x 16B async copies per thread (8 KB total) ----
        const __bf16* ag = A + (size_t)(m0 + ar) * Kdim + (k0 + ac);
        async_ld_b128_(asA0, ag);
        async_ld_b128_(asA1, ag + 8);

        // ---- B tile: f32 load, cvt, transposed LDS store ----
        const float* wg = W + (size_t)(k0 + bk) * N + (n0 + bc);
        if (k0 + 32 < Kdim)
            __builtin_prefetch(wg + (size_t)32 * N, 0, 3);  // global_prefetch_b8
        #pragma unroll
        for (int j = 0; j < 8; ++j) Bs[bc + j][bk] = (__bf16)wg[j];

        wait_asynccnt0_();
        __syncthreads();

        // A fragment: lane (hf,l16): row = l16; k = hf*16 + e
        bf16x16 afrag[2], bfrag[2];
        #pragma unroll
        for (int ms = 0; ms < 2; ++ms) {
            const int row = wm * 32 + ms * 16 + l16;
            #pragma unroll
            for (int e = 0; e < 16; ++e) {
                const int kk = hf * 16 + e;
                afrag[ms][e] = As[row][kk];
            }
        }
        // B fragment: N = l16, k = hf*16 + e (contiguous in transposed Bs)
        #pragma unroll
        for (int ns = 0; ns < 2; ++ns) {
            const int col = wn * 32 + ns * 16 + l16;
            #pragma unroll
            for (int e = 0; e < 16; ++e)
                bfrag[ns][e] = Bs[col][hf * 16 + e];
        }

        #pragma unroll
        for (int ms = 0; ms < 2; ++ms)
            #pragma unroll
            for (int ns = 0; ns < 2; ++ns)
                acc[ms][ns] = __builtin_amdgcn_wmma_f32_16x16x32_bf16(
                    false, afrag[ms], false, bfrag[ns],
                    (short)0, acc[ms][ns], false, false);
        __syncthreads();
    }

    // epilogue: C/D layout -> row = j + 8*hf, col = l16
    #pragma unroll
    for (int ms = 0; ms < 2; ++ms)
        #pragma unroll
        for (int ns = 0; ns < 2; ++ns)
            #pragma unroll
            for (int j = 0; j < 8; ++j) {
                const int rowg = m0 + wm * 32 + ms * 16 + j + hf * 8;
                const int colg = n0 + wn * 32 + ns * 16 + l16;
                C[(size_t)rowg * N + colg] = acc[ms][ns][j];
            }
}

// ---------------------------------------------------------------------------
// Kernel 2: beta = sigmoid(x@Wb); g = -exp(A_log)*softplus(x@Wgk + dt_bias).
// One block per (b,t) row.  32 outputs, 8 partial threads each.
// ---------------------------------------------------------------------------
__global__ __launch_bounds__(256)
void betag_kernel(const float* __restrict__ x,
                  const float* __restrict__ Wb, const float* __restrict__ Wgk,
                  const float* __restrict__ A_log, const float* __restrict__ dtb,
                  float* __restrict__ beta, float* __restrict__ g) {
    __shared__ float red[32][8];
    const int row = blockIdx.x;                 // 0..ML_-1
    const float* xr = x + (size_t)row * DH_;
    const int o = threadIdx.x & 31;             // output 0..31
    const int p = threadIdx.x >> 5;             // partial 0..7
    const float* Wcol = (o < 16) ? (Wb + o) : (Wgk + (o - 16));
    float s = 0.f;
    const int k0 = p * (DH_ / 8);
    for (int k = k0; k < k0 + DH_ / 8; ++k)
        s += xr[k] * Wcol[(size_t)k * 16];
    red[o][p] = s;
    __syncthreads();
    if (threadIdx.x < 32) {
        float t = 0.f;
        #pragma unroll
        for (int j = 0; j < 8; ++j) t += red[threadIdx.x][j];
        if (threadIdx.x < 16) {
            beta[(size_t)row * H_ + threadIdx.x] = sigmoidf_(t);
        } else {
            const int h = threadIdx.x - 16;
            const float z  = t + dtb[h];
            const float sp = (z > 20.f) ? z : log1pf(__expf(z));
            g[(size_t)row * H_ + h] = -__expf(A_log[h]) * sp;
        }
    }
}

// ---------------------------------------------------------------------------
// Kernel 3: causal depthwise conv (K=4) + SiLU + l2norm + GQA head expansion.
// One block per (b,t,h); 128 threads = one d each.  Writes qn/kn/vn [bt,h,d].
// ---------------------------------------------------------------------------
__device__ __forceinline__ float block_sumsq128(float v, float* sbuf) {
    float s = v * v;
    #pragma unroll
    for (int off = 16; off > 0; off >>= 1) s += __shfl_xor(s, off, 32);
    const int lane = threadIdx.x & 31, w = threadIdx.x >> 5;
    if (lane == 0) sbuf[w] = s;
    __syncthreads();
    const float tot = sbuf[0] + sbuf[1] + sbuf[2] + sbuf[3];
    __syncthreads();
    return tot;
}

__global__ __launch_bounds__(128)
void conv_norm_kernel(const float* __restrict__ qp, const float* __restrict__ kp,
                      const float* __restrict__ vp,
                      const float* __restrict__ cq, const float* __restrict__ ck,
                      const float* __restrict__ cv,
                      float* __restrict__ qn, float* __restrict__ kn,
                      float* __restrict__ vn) {
    __shared__ float sbuf[4];
    const int idx = blockIdx.x;          // bt*H + h
    const int h   = idx % H_;
    const int bt  = idx / H_;
    const int t   = bt % L_;
    const int d   = threadIdx.x;
    const size_t obase = (size_t)idx * D_;

    // ---- q: channel h*D+d in [bt, 2048] ----
    {
        const int c = h * D_ + d;
        float a = 0.f;
        #pragma unroll
        for (int j = 0; j < KC_; ++j) {
            const int tt = t - (KC_ - 1) + j;
            if (tt >= 0)
                a += qp[(size_t)(bt - (KC_ - 1) + j) * (HQ_ * D_) + c] * cq[c * KC_ + j];
        }
        const float y  = siluf_(a);
        const float ss = block_sumsq128(y, sbuf);
        qn[obase + d]  = y * rsqrtf(ss + 1e-6f) * 0.08838834764831845f; // D^-0.5
    }
    // ---- k: GQA source head h/4 ----
    {
        const int c = (h >> 2) * D_ + d;
        float a = 0.f;
        #pragma unroll
        for (int j = 0; j < KC_; ++j) {
            const int tt = t - (KC_ - 1) + j;
            if (tt >= 0)
                a += kp[(size_t)(bt - (KC_ - 1) + j) * (HK_ * D_) + c] * ck[c * KC_ + j];
        }
        const float y  = siluf_(a);
        const float ss = block_sumsq128(y, sbuf);
        kn[obase + d]  = y * rsqrtf(ss + 1e-6f);
    }
    // ---- v: conv+silu only ----
    {
        const int c = (h >> 2) * D_ + d;
        float a = 0.f;
        #pragma unroll
        for (int j = 0; j < KC_; ++j) {
            const int tt = t - (KC_ - 1) + j;
            if (tt >= 0)
                a += vp[(size_t)(bt - (KC_ - 1) + j) * (HV_ * D_) + c] * cv[c * KC_ + j];
        }
        vn[obase + d] = siluf_(a);
    }
}

// ---------------------------------------------------------------------------
// Kernel 4: gated delta-rule recurrence.  One block per (b,h) chain.
// 256 threads: thread (hf,dv) owns S[dk in hf*64..+63][dv] in 64 VGPRs.
//   S = exp(g)*S;  vres = v - k.S;  S += beta*k (x) vres;  o = q.S
// ---------------------------------------------------------------------------
__global__ __launch_bounds__(256)
void recurrence_kernel(const float* __restrict__ qn, const float* __restrict__ kn,
                       const float* __restrict__ vn, const float* __restrict__ gd,
                       const float* __restrict__ beta, float* __restrict__ ob) {
    const int bh = blockIdx.x;             // b*H + h
    const int b  = bh / H_;
    const int h  = bh % H_;
    const int dv = threadIdx.x & 127;
    const int hf = threadIdx.x >> 7;       // 0/1 : which half of dk
    const int dk0 = hf * 64;

    __shared__ float kst[128], qst[128], vst[128];
    __shared__ float part[256];
    __shared__ float vres[128];

    float S[64];
    #pragma unroll
    for (int i = 0; i < 64; ++i) S[i] = 0.f;

    for (int t = 0; t < L_; ++t) {
        const size_t base = ((size_t)(b * L_ + t) * H_ + h) * D_;
        if (hf == 0) { kst[dv] = kn[base + dv]; vst[dv] = vn[base + dv]; }
        else         { qst[dv] = qn[base + dv]; }
        __syncthreads();

        const size_t sidx = (size_t)(b * L_ + t) * H_ + h;
        const float eg = __expf(gd[sidx]);
        const float bt = beta[sidx];

        float kv = 0.f;
        #pragma unroll
        for (int i = 0; i < 64; ++i) {
            S[i] *= eg;
            kv += kst[dk0 + i] * S[i];
        }
        part[threadIdx.x] = kv;
        __syncthreads();
        if (hf == 0) vres[dv] = vst[dv] - (part[dv] + part[dv + 128]);
        __syncthreads();

        const float vr = vres[dv];
        float o = 0.f;
        #pragma unroll
        for (int i = 0; i < 64; ++i) {
            S[i] += bt * kst[dk0 + i] * vr;
            o += qst[dk0 + i] * S[i];
        }
        part[threadIdx.x] = o;
        __syncthreads();
        if (hf == 0) ob[base + dv] = part[dv] + part[dv + 128];
        __syncthreads();
    }
}

// ---------------------------------------------------------------------------
// Kernel 5: oo = rmsnorm(o) * w * silu(gate)  -> bf16 (A operand of final GEMM)
// ---------------------------------------------------------------------------
__global__ __launch_bounds__(128)
void gated_norm_kernel(const float* __restrict__ ob, const float* __restrict__ gatep,
                       const float* __restrict__ w, __bf16* __restrict__ oo) {
    __shared__ float sbuf[4];
    const int idx = blockIdx.x;            // bt*H + h  (flat col = h*D+d)
    const int d   = threadIdx.x;
    const size_t fi = (size_t)idx * D_ + d;
    const float o  = ob[fi];
    const float ss = block_sumsq128(o, sbuf);
    const float rs = rsqrtf(ss * (1.0f / D_) + 1e-5f);
    const float gt = gatep[fi];
    oo[fi] = (__bf16)(o * rs * w[d] * siluf_(gt));
}

// ---------------------------------------------------------------------------
// Host-side launcher (graph-capture safe: kernel launches only)
// ---------------------------------------------------------------------------
extern "C" void kernel_launch(void* const* d_in, const int* in_sizes, int n_in,
                              void* d_out, int out_size, void* d_ws, size_t ws_size,
                              hipStream_t stream) {
    const float* x     = (const float*)d_in[0];
    const float* Wq    = (const float*)d_in[1];
    const float* Wk    = (const float*)d_in[2];
    const float* Wv    = (const float*)d_in[3];
    const float* Wb    = (const float*)d_in[4];
    const float* Wgk   = (const float*)d_in[5];
    const float* A_log = (const float*)d_in[6];
    const float* dtb   = (const float*)d_in[7];
    const float* cq    = (const float*)d_in[8];
    const float* ck    = (const float*)d_in[9];
    const float* cv    = (const float*)d_in[10];
    const float* Wg    = (const float*)d_in[11];
    const float* onw   = (const float*)d_in[12];
    const float* Wo    = (const float*)d_in[13];
    float* out = (float*)d_out;

    // workspace layout (f32 slots)
    float* ws = (float*)d_ws;
    size_t off = 0;
    float* qproj = ws + off; off += (size_t)ML_ * (HQ_ * D_);  // 8.39M
    float* kproj = ws + off; off += (size_t)ML_ * (HK_ * D_);  // 2.10M
    float* vproj = ws + off; off += (size_t)ML_ * (HV_ * D_);  // 2.10M
    float* gatep = ws + off; off += (size_t)ML_ * HD_;         // 8.39M
    float* betab = ws + off; off += (size_t)ML_ * H_;
    float* gbuf  = ws + off; off += (size_t)ML_ * H_;
    float* qn    = ws + off; off += (size_t)ML_ * HD_;
    float* kn    = ws + off; off += (size_t)ML_ * HD_;
    float* vn    = ws + off; off += (size_t)ML_ * HD_;
    __bf16* xbf  = (__bf16*)(ws + off); off += (size_t)ML_ * DH_ / 2;  // bf16 copy of x
    float* ob    = qproj;              // recurrence output reuses qproj (consumed)
    __bf16* oobf = (__bf16*)kn;        // gated-norm bf16 output reuses kn (consumed)

    const dim3 blk256(256), blk128(128);
    const int nX = ML_ * DH_;

    // 0) bf16 copy of hidden states (A operand of the 4 projection GEMMs)
    to_bf16_kernel<<<dim3(nX / 1024), blk256, 0, stream>>>(x, xbf, nX);

    // 1) projections (async-LDS staged bf16 WMMA GEMMs)
    wmma_gemm_kernel<<<dim3((HQ_ * D_) / 64, ML_ / 128), blk256, 0, stream>>>(
        xbf, Wq, qproj, ML_, HQ_ * D_, DH_);
    wmma_gemm_kernel<<<dim3((HK_ * D_) / 64, ML_ / 128), blk256, 0, stream>>>(
        xbf, Wk, kproj, ML_, HK_ * D_, DH_);
    wmma_gemm_kernel<<<dim3((HV_ * D_) / 64, ML_ / 128), blk256, 0, stream>>>(
        xbf, Wv, vproj, ML_, HV_ * D_, DH_);
    wmma_gemm_kernel<<<dim3(HD_ / 64, ML_ / 128), blk256, 0, stream>>>(
        xbf, Wg, gatep, ML_, HD_, DH_);

    // 2) beta / g scalars
    betag_kernel<<<dim3(ML_), blk256, 0, stream>>>(x, Wb, Wgk, A_log, dtb,
                                                   betab, gbuf);

    // 3) conv + silu + l2norm + GQA expansion
    conv_norm_kernel<<<dim3(ML_ * H_), blk128, 0, stream>>>(
        qproj, kproj, vproj, cq, ck, cv, qn, kn, vn);

    // 4) gated delta-rule recurrence (one block per (b,h) chain)
    recurrence_kernel<<<dim3(B_ * H_), blk256, 0, stream>>>(
        qn, kn, vn, gbuf, betab, ob);

    // 5) gated RMSNorm -> bf16
    gated_norm_kernel<<<dim3(ML_ * H_), blk128, 0, stream>>>(ob, gatep, onw, oobf);

    // 6) output projection -> d_out
    wmma_gemm_kernel<<<dim3(DH_ / 64, ML_ / 128), blk256, 0, stream>>>(
        oobf, Wo, out, ML_, DH_, HD_);

    (void)in_sizes; (void)n_in; (void)out_size; (void)ws_size;
}
